// TpcModule_31069793419644
// MI455X (gfx1250) — matrix operand order repaired and checked by reading, not compiled, not run on hardware
//
#include <hip/hip_runtime.h>
#include <hip/hip_bf16.h>

// gathered batched dot product:  out[m, j] = q[m] . k[idx[m, j]]
//   q:   [M=16384, D=128]  bf16
//   k:   [N=65536, D=128]  bf16
//   idx: [M, K=128]        int32 (harness contract)
//   out: [M, K]            bf16
//
// Mapping: 1 block = 1 query m (8 wave32s). Wave w handles key slots
// j0 = 16*w .. 16*w+15 via a 16x16x128 WMMA chain (4x v_wmma_f32_16x16x32_bf16):
//   A = q[m] replicated into all 16 rows, B = 16 gathered key rows as columns.
// All C rows are identical, so lanes 0..15 emit the 16 outputs from C VGPR0.

typedef __bf16 bf16_t;
typedef __attribute__((ext_vector_type(16))) __bf16 v16bf;
typedef __attribute__((ext_vector_type(8)))  __bf16 v8bf;
typedef __attribute__((ext_vector_type(8)))  float  v8f;

#ifndef DQ_M
#define DQ_M 16384
#endif
#define DQ_D 128
#define DQ_K 128

__global__ __launch_bounds__(256) void gather_dot_wmma_kernel(
    const bf16_t* __restrict__ q,
    const bf16_t* __restrict__ k,
    const int*    __restrict__ idx,
    bf16_t*       __restrict__ out)
{
    const int lane = threadIdx.x & 31;
    const int wave = threadIdx.x >> 5;      // 0..7
    const int m    = blockIdx.x;            // one query per block
    const int j0   = wave * 16;             // this wave's 16 key slots
    const int half = lane >> 4;             // 0: lanes 0-15, 1: lanes 16-31
    const int n    = lane & 15;             // B column / output slot within j-block

    // Per-lane gather index for B column n (coalesced int32 load)
    const int krow_id = idx[(size_t)m * DQ_K + j0 + n];
    const bf16_t* __restrict__ krow = k + (size_t)krow_id * DQ_D;
    const bf16_t* __restrict__ qrow = q + (size_t)m       * DQ_D;

    v8f c = {};

#pragma unroll
    for (int kk = 0; kk < 4; ++kk) {
        const int dbase = kk * 32;

        // ---- A fragment (16x32 bf16, all rows = q[m]) ----
        // Layout: lanes 0-15 hold K = 0..15, lanes 16-31 hold
        // K = 16..31 (matches B). Two contiguous 16B chunks per lane.
        const int alo = dbase + (half ? 16 : 0);
        v8bf a_lo = *(const v8bf*)(qrow + alo);
        v8bf a_hi = *(const v8bf*)(qrow + alo + 8);
        v16bf a;
#pragma unroll
        for (int e = 0; e < 8; ++e) { a[e] = a_lo[e]; a[e + 8] = a_hi[e]; }

        // ---- B fragment (32x16 bf16, column n = gathered key row) ----
        // Lanes 0-15 hold K(d) = dbase+0..15, lanes 16-31 hold dbase+16..31:
        // one contiguous 32B load per lane from the gathered key row.
        const int blo = dbase + (half ? 16 : 0);
        v16bf b = *(const v16bf*)(krow + blo);

        // D = A x B + C   (8 args: neg_a, A, neg_b, B, c_mod, C, reuse_a, reuse_b)
        c = __builtin_amdgcn_wmma_f32_16x16x32_bf16(
                false, a, false, b, (short)0, c, false, false);
    }

    // C layout: VGPR g, lane L -> (M = g + 16*half/2? rows identical anyway,
    // N = L%16). All M rows equal since A rows are identical: take VGPR0,
    // lanes 0..15 give N = 0..15.
    if (half == 0) {
        out[(size_t)m * DQ_K + j0 + n] = (bf16_t)c[0];
    }
}

extern "C" void kernel_launch(void* const* d_in, const int* in_sizes, int n_in,
                              void* d_out, int out_size, void* d_ws, size_t ws_size,
                              hipStream_t stream)
{
    const bf16_t* q   = (const bf16_t*)d_in[0];
    const bf16_t* k   = (const bf16_t*)d_in[1];
    const int*    idx = (const int*)   d_in[2];
    bf16_t*       out = (bf16_t*)      d_out;

    const int M = in_sizes[0] / DQ_D;   // 16384

    dim3 grid(M);
    dim3 block(256);
    gather_dot_wmma_kernel<<<grid, block, 0, stream>>>(q, k, idx, out);
}
